// EnhancedGEARS_74079595922009
// MI455X (gfx1250) — compile-verified, matrix operand order checked by reading, not run on hardware
//
#include <hip/hip_runtime.h>
#include <hip/hip_bf16.h>

// ---------------------------------------------------------------------------
// EnhancedGEARS forward for MI455X (gfx1250, wave32, WMMA).
// All GEMMs go through v_wmma_f32_16x16x32_f16 (f16 inputs, f32 accumulate).
// Elementwise / LN / GAT-softmax / scatter kernels are plain VALU (mem-bound).
// ---------------------------------------------------------------------------

typedef __attribute__((ext_vector_type(4)))  float    v4f;
typedef __attribute__((ext_vector_type(8)))  float    v8f;
typedef __attribute__((ext_vector_type(4)))  _Float16 v4h;
typedef __attribute__((ext_vector_type(8)))  _Float16 v8h;
typedef __attribute__((ext_vector_type(16))) _Float16 v16h;

#define N_NODES 8192
#define D_IN    2048
#define H_DIM   1024
#define HEADS   8
#define CH      128
#define N_EDGES 65536
#define E_TOT   (N_EDGES + N_NODES)   // edges + self loops

// scheduling pins: group all DS reads before the WMMA chain
#if __has_builtin(__builtin_amdgcn_sched_group_barrier)
#define SCHED_GROUP(mask, size) __builtin_amdgcn_sched_group_barrier((mask), (size), 0)
#else
#define SCHED_GROUP(mask, size)
#endif
#define SG_DS_READ 0x100
#define SG_MATRIX  0x008

// ---------------------------------------------------------------------------
// Tiled WMMA GEMM:  C[M x Nout] = A[M x K] * W[Nout x K]^T (+ bias) (+ beta*C)
// Block = 256 threads (8 waves).  Tile = 128x128, K-step = 64 (2 WMMA halves).
// Wave w computes rows [w*16, w*16+16) x 128 cols = 8 accumulator tiles.
// A and W are both K-major -> identical LDS fragment layout for both.
// Schedule per half is pinned to: 18x ds_load_b128, then 8x back-to-back
// v_wmma (single dscnt drain per half instead of one per WMMA).
// ---------------------------------------------------------------------------
__global__ __launch_bounds__(256)
void gemm_wmma_f16(const float* __restrict__ A, int lda,
                   const float* __restrict__ W, int ldb,
                   float* __restrict__ C, int ldc,
                   const float* __restrict__ bias,
                   int K, int beta)
{
    __shared__ _Float16 As[128][72];   // 64 cols + pad -> 144B rows (16B aligned)
    __shared__ _Float16 Bs[128][72];

    const int t    = threadIdx.x;
    const int wave = t >> 5;
    const int lane = t & 31;
    const int rowBase = blockIdx.y * 128;
    const int colBase = blockIdx.x * 128;

    v8f zero = {};
    v8f acc[8];
#pragma unroll
    for (int j = 0; j < 8; ++j) acc[j] = zero;

    // cooperative staging: 16 threads per row (64 floats), 16 rows per pass
    const int lr = t >> 4;          // 0..15
    const int lc = (t & 15) << 2;   // 0,4,...,60
    const float* Ap = A + (size_t)(rowBase + lr) * lda + lc;
    const float* Wp = W + (size_t)(colBase + lr) * ldb + lc;

    const int m  = wave * 16 + (lane & 15);
    const int kh = (lane < 16) ? 0 : 8;

    for (int k0 = 0; k0 < K; k0 += 64) {
        // --- stage 128x64 tiles of A and W into LDS as f16 ---
#pragma unroll
        for (int rr = 0; rr < 128; rr += 16) {
            v4f av = *(const v4f*)(Ap + (size_t)rr * lda + k0);
            v4f wv = *(const v4f*)(Wp + (size_t)rr * ldb + k0);
            v4h ah = { (_Float16)av[0], (_Float16)av[1], (_Float16)av[2], (_Float16)av[3] };
            v4h wh = { (_Float16)wv[0], (_Float16)wv[1], (_Float16)wv[2], (_Float16)wv[3] };
            *(v4h*)&As[rr + lr][lc] = ah;
            *(v4h*)&Bs[rr + lr][lc] = wh;
        }
        __syncthreads();

        // prefetch next K-tile (global_prefetch_b8 path)
        if (k0 + 64 < K) {
            int r = t & 127;
            __builtin_prefetch(A + (size_t)(rowBase + r) * lda + k0 + 64, 0, 1);
            __builtin_prefetch(W + (size_t)(colBase + r) * ldb + k0 + 64, 0, 1);
        }

        // --- two K=32 WMMA halves per staged tile ---
        // Fragment layout (ISA 16-bit A/B): lane 0-15: M/N=lane,
        //   elems 0..7 = K kb..kb+7, elems 8..15 = K kb+16..kb+23
        // lane 16-31: M/N=lane-16, halves shifted by 8.
#pragma unroll
        for (int half = 0; half < 2; ++half) {
            const int kb = half * 32 + kh;
            v8h alo = *(const v8h*)&As[m][kb];
            v8h ahi = *(const v8h*)&As[m][kb + 16];
            v16h af = __builtin_shufflevector(alo, ahi,
                       0,1,2,3,4,5,6,7,8,9,10,11,12,13,14,15);

            v16h bf[8];
#pragma unroll
            for (int j = 0; j < 8; ++j) {
                int n = j * 16 + (lane & 15);
                v8h blo = *(const v8h*)&Bs[n][kb];
                v8h bhi = *(const v8h*)&Bs[n][kb + 16];
                bf[j] = __builtin_shufflevector(blo, bhi,
                         0,1,2,3,4,5,6,7,8,9,10,11,12,13,14,15);
            }
#pragma unroll
            for (int j = 0; j < 8; ++j)
                acc[j] = __builtin_amdgcn_wmma_f32_16x16x32_f16(
                             false, af, false, bf[j], (short)0, acc[j], false, false);

            // pin: all 18 LDS reads of this half first, then the 8 WMMAs
            SCHED_GROUP(SG_DS_READ, 18);
            SCHED_GROUP(SG_MATRIX,   8);
        }
        __syncthreads();
    }

    // --- epilogue: C-tile layout VGPR r -> M = r (lanes 0-15) / 8+r (16-31) ---
    const int rbase = rowBase + wave * 16 + ((lane < 16) ? 0 : 8);
#pragma unroll
    for (int j = 0; j < 8; ++j) {
        int col = colBase + j * 16 + (lane & 15);
        float bv = bias ? bias[col] : 0.f;
#pragma unroll
        for (int r = 0; r < 8; ++r) {
            size_t off = (size_t)(rbase + r) * ldc + col;
            float v = acc[j][r] + bv;
            if (beta) v += C[off];
            C[off] = v;
        }
    }
}

// ---------------------------------------------------------------------------
// LayerNorm(1024) + ReLU, in place.  One block (256 thr) per row.
// ---------------------------------------------------------------------------
__global__ void ln_relu_k(float* __restrict__ X,
                          const float* __restrict__ g,
                          const float* __restrict__ be)
{
    __shared__ float s1[256], s2[256];
    const int row = blockIdx.x, t = threadIdx.x;
    float* xp = X + (size_t)row * H_DIM + t * 4;
    v4f v = *(v4f*)xp;
    s1[t] = v[0] + v[1] + v[2] + v[3];
    s2[t] = v[0]*v[0] + v[1]*v[1] + v[2]*v[2] + v[3]*v[3];
    __syncthreads();
    for (int o = 128; o > 0; o >>= 1) {
        if (t < o) { s1[t] += s1[t + o]; s2[t] += s2[t + o]; }
        __syncthreads();
    }
    const float mu  = s1[0] * (1.f / H_DIM);
    const float var = s2[0] * (1.f / H_DIM) - mu * mu;
    const float rs  = rsqrtf(var + 1e-5f);
    v4f gv = *(const v4f*)(g  + t * 4);
    v4f bv = *(const v4f*)(be + t * 4);
    v4f o4;
#pragma unroll
    for (int i = 0; i < 4; ++i)
        o4[i] = fmaxf((v[i] - mu) * rs * gv[i] + bv[i], 0.f);
    *(v4f*)xp = o4;
}

__global__ void relu_row_k(float* __restrict__ X)
{
    float* xp = X + (size_t)blockIdx.x * H_DIM + threadIdx.x * 4;
    v4f v = *(v4f*)xp;
#pragma unroll
    for (int i = 0; i < 4; ++i) v[i] = fmaxf(v[i], 0.f);
    *(v4f*)xp = v;
}

// ---------------------------------------------------------------------------
// GAT pieces
// ---------------------------------------------------------------------------
__device__ inline unsigned fenc(float f) {            // order-preserving f32->u32
    unsigned u = __float_as_uint(f);
    return u ^ (((int)u >> 31) | 0x80000000u);
}
__device__ inline float fdec(unsigned u) {
    unsigned m = (u >> 31) ? 0x80000000u : 0xFFFFFFFFu;
    return __uint_as_float(u ^ m);
}

// per-node attention coefficients: asrc/adst[n,h] = dot(xw[n,h,:], a_src/dst[h,:])
__global__ void gat_coeff_k(const float* __restrict__ xw,
                            const float* __restrict__ a_src,
                            const float* __restrict__ a_dst,
                            float* __restrict__ asrc, float* __restrict__ adst)
{
    const int n = blockIdx.x, t = threadIdx.x;
    const int h = t >> 5, lane = t & 31;
    v4f xv = *(const v4f*)(xw + (size_t)n * H_DIM + h * CH + lane * 4);
    v4f sv = *(const v4f*)(a_src + h * CH + lane * 4);
    v4f dv = *(const v4f*)(a_dst + h * CH + lane * 4);
    float s = xv[0]*sv[0] + xv[1]*sv[1] + xv[2]*sv[2] + xv[3]*sv[3];
    float d = xv[0]*dv[0] + xv[1]*dv[1] + xv[2]*dv[2] + xv[3]*dv[3];
#pragma unroll
    for (int o = 16; o > 0; o >>= 1) {
        s += __shfl_down(s, o, 32);
        d += __shfl_down(d, o, 32);
    }
    if (lane == 0) { asrc[n * HEADS + h] = s; adst[n * HEADS + h] = d; }
}

__global__ void gat_init_k(unsigned* __restrict__ menc, float* __restrict__ denom)
{
    int i = blockIdx.x * blockDim.x + threadIdx.x;
    if (i < N_NODES * HEADS) { menc[i] = 0x007FFFFFu /* enc(-inf) */; denom[i] = 0.f; }
}

__global__ void gat_edge_max_k(const int* __restrict__ ei,
                               const float* __restrict__ asrc,
                               const float* __restrict__ adst,
                               float* __restrict__ alpha,
                               unsigned* __restrict__ menc)
{
    int gid = blockIdx.x * blockDim.x + threadIdx.x;
    if (gid >= E_TOT * HEADS) return;
    int e = gid >> 3, h = gid & 7;
    int s, d;
    if (e < N_EDGES) { s = ei[e]; d = ei[N_EDGES + e]; } else { s = d = e - N_EDGES; }
    float a = asrc[s * HEADS + h] + adst[d * HEADS + h];
    a = (a > 0.f) ? a : 0.2f * a;                 // leaky_relu
    alpha[gid] = a;
    atomicMax(&menc[d * HEADS + h], fenc(a));
}

__global__ void gat_edge_exp_k(const int* __restrict__ ei,
                               float* __restrict__ alpha,    // in: logits, out: exp
                               const unsigned* __restrict__ menc,
                               float* __restrict__ denom)
{
    int gid = blockIdx.x * blockDim.x + threadIdx.x;
    if (gid >= E_TOT * HEADS) return;
    int e = gid >> 3, h = gid & 7;
    int d = (e < N_EDGES) ? ei[N_EDGES + e] : (e - N_EDGES);
    float ex = __expf(alpha[gid] - fdec(menc[d * HEADS + h]));
    alpha[gid] = ex;
    atomicAdd(&denom[d * HEADS + h], ex);
}

__global__ void gat_out_init_k(float* __restrict__ out, const float* __restrict__ bias)
{
    v4f b4 = *(const v4f*)(bias + threadIdx.x * 4);
    *(v4f*)(out + (size_t)blockIdx.x * H_DIM + threadIdx.x * 4) = b4;
}

// weighted scatter-add: out[dst] += xw[src] * (ex/denom[dst]).  1 block per edge.
__global__ void gat_scatter_k(const int* __restrict__ ei,
                              const float* __restrict__ xw,
                              const float* __restrict__ ex,
                              const float* __restrict__ denom,
                              float* __restrict__ out)
{
    const int e = blockIdx.x, t = threadIdx.x;
    int s, d;
    if (e < N_EDGES) { s = ei[e]; d = ei[N_EDGES + e]; } else { s = d = e - N_EDGES; }
    const int h = t >> 5;   // col = t*4 .. t*4+3 all inside head h (128-wide)
    float w = ex[(size_t)e * HEADS + h] / denom[d * HEADS + h];
    v4f xv = *(const v4f*)(xw + (size_t)s * H_DIM + t * 4);
    float* op = out + (size_t)d * H_DIM + t * 4;
#pragma unroll
    for (int i = 0; i < 4; ++i) atomicAdd(op + i, xv[i] * w);
}

// ---------------------------------------------------------------------------
// Host launcher
// ---------------------------------------------------------------------------
extern "C" void kernel_launch(void* const* d_in, const int* in_sizes, int n_in,
                              void* d_out, int out_size, void* d_ws, size_t ws_size,
                              hipStream_t stream)
{
    (void)in_sizes; (void)n_in; (void)out_size; (void)ws_size;

    const float* x    = (const float*)d_in[0];
    const float* pert = (const float*)d_in[1];
    // d_in[2] = edge_weight (ignored by the reference: edge_dim=None)
    // params in dict-insertion order starting at index 3:
    const float* eeW1 = (const float*)d_in[3];  const float* eeb1 = (const float*)d_in[4];
    const float* eeg1 = (const float*)d_in[5];  const float* eebe1= (const float*)d_in[6];
    const float* eeW2 = (const float*)d_in[7];  const float* eeb2 = (const float*)d_in[8];
    const float* eeg2 = (const float*)d_in[9];  const float* eebe2= (const float*)d_in[10];
    const float* peW1 = (const float*)d_in[11]; const float* peb1 = (const float*)d_in[12];
    const float* peg1 = (const float*)d_in[13]; const float* pebe1= (const float*)d_in[14];
    const float* peW2 = (const float*)d_in[15]; const float* peb2 = (const float*)d_in[16];
    const float* peg2 = (const float*)d_in[17]; const float* pebe2= (const float*)d_in[18];
    const float* Wv   = (const float*)d_in[19]; const float* bv   = (const float*)d_in[20];
    const float* Wo   = (const float*)d_in[21]; const float* bo   = (const float*)d_in[22];
    const float* g0W  = (const float*)d_in[23]; const float* g0as = (const float*)d_in[24];
    const float* g0ad = (const float*)d_in[25]; const float* g0b  = (const float*)d_in[26];
    const float* g1W  = (const float*)d_in[27]; const float* g1as = (const float*)d_in[28];
    const float* g1ad = (const float*)d_in[29]; const float* g1b  = (const float*)d_in[30];
    const float* ceW1 = (const float*)d_in[31]; const float* ceb1 = (const float*)d_in[32];
    const float* ceg1 = (const float*)d_in[33]; const float* cebe1= (const float*)d_in[34];
    const float* ceW2 = (const float*)d_in[35]; const float* ceb2 = (const float*)d_in[36];
    const float* ceg2 = (const float*)d_in[37]; const float* cebe2= (const float*)d_in[38];
    const float* fuW1 = (const float*)d_in[39]; const float* fub1 = (const float*)d_in[40];
    const float* fug1 = (const float*)d_in[41]; const float* fube1= (const float*)d_in[42];
    const float* fuW2 = (const float*)d_in[43]; const float* fub2 = (const float*)d_in[44];
    const float* fug2 = (const float*)d_in[45]; const float* fube2= (const float*)d_in[46];
    const float* oW1  = (const float*)d_in[47]; const float* ob1  = (const float*)d_in[48];
    const float* og1  = (const float*)d_in[49]; const float* obe1 = (const float*)d_in[50];
    const float* oW2  = (const float*)d_in[51]; const float* ob2  = (const float*)d_in[52];
    const int*   ei   = (const int*)d_in[53];   // [2, E] flattened
    float* out = (float*)d_out;

    // workspace carve-up: 6 x [8192,1024] f32 + small GAT buffers
    const size_t S = (size_t)N_NODES * H_DIM * sizeof(float);   // 32 MB
    char* ws = (char*)d_ws;
    float* bA = (float*)(ws + 0 * S);   // generic scratch / MLP hidden
    float* bB = (float*)(ws + 1 * S);   // gene ping
    float* bC = (float*)(ws + 2 * S);   // gene pong
    float* bP = (float*)(ws + 3 * S);   // pert features
    float* bX = (float*)(ws + 4 * S);   // GAT xw / fused
    float* bM = (float*)(ws + 5 * S);   // comb
    float*    asrc  = (float*)(ws + 6 * S);
    float*    adst  = asrc  + (size_t)N_NODES * HEADS;
    unsigned* menc  = (unsigned*)(adst + (size_t)N_NODES * HEADS);
    float*    denom = (float*)(menc + (size_t)N_NODES * HEADS);
    float*    exbuf = denom + (size_t)N_NODES * HEADS;          // [E_TOT, HEADS]

    auto gemm = [&](const float* A, int lda, const float* W, int ldb,
                    float* C, int ldc, const float* bias, int Nout, int K, int beta) {
        dim3 grid(Nout / 128, N_NODES / 128);
        gemm_wmma_f16<<<grid, 256, 0, stream>>>(A, lda, W, ldb, C, ldc, bias, K, beta);
    };
    auto ln = [&](float* X, const float* g, const float* b) {
        ln_relu_k<<<N_NODES, 256, 0, stream>>>(X, g, b);
    };
    auto run_gat = [&](const float* Wg, const float* a_s, const float* a_d,
                       const float* bg, const float* gene_in, float* gene_out) {
        gemm(gene_in, H_DIM, Wg, H_DIM, bX, H_DIM, nullptr, H_DIM, H_DIM, 0);
        gat_coeff_k<<<N_NODES, 256, 0, stream>>>(bX, a_s, a_d, asrc, adst);
        gat_init_k<<<(N_NODES * HEADS + 255) / 256, 256, 0, stream>>>(menc, denom);
        gat_edge_max_k<<<(E_TOT * HEADS + 255) / 256, 256, 0, stream>>>(ei, asrc, adst, exbuf, menc);
        gat_edge_exp_k<<<(E_TOT * HEADS + 255) / 256, 256, 0, stream>>>(ei, exbuf, menc, denom);
        gat_out_init_k<<<N_NODES, 256, 0, stream>>>(gene_out, bg);
        gat_scatter_k<<<E_TOT, 256, 0, stream>>>(ei, bX, exbuf, denom, gene_out);
        relu_row_k<<<N_NODES, 256, 0, stream>>>(gene_out);
    };

    // --- expression encoder ---
    gemm(x, D_IN, eeW1, D_IN, bA, H_DIM, eeb1, H_DIM, D_IN, 0);  ln(bA, eeg1, eebe1);
    gemm(bA, H_DIM, eeW2, H_DIM, bB, H_DIM, eeb2, H_DIM, H_DIM, 0); ln(bB, eeg2, eebe2);
    // --- pert encoder ---
    gemm(pert, D_IN, peW1, D_IN, bA, H_DIM, peb1, H_DIM, D_IN, 0); ln(bA, peg1, pebe1);
    gemm(bA, H_DIM, peW2, H_DIM, bP, H_DIM, peb2, H_DIM, H_DIM, 0); ln(bP, peg2, pebe2);
    // --- attention with seq_len==1: out = Wo(Wv x + bv) + bo ---
    gemm(bB, H_DIM, Wv, H_DIM, bA, H_DIM, bv, H_DIM, H_DIM, 0);
    gemm(bA, H_DIM, Wo, H_DIM, bC, H_DIM, bo, H_DIM, H_DIM, 0);
    // --- two GAT layers (ping-pong gene buffers) ---
    run_gat(g0W, g0as, g0ad, g0b, bC, bB);
    run_gat(g1W, g1as, g1ad, g1b, bB, bC);
    // --- combination encoder on concat[gene, pertf] via split-K ---
    gemm(bC, H_DIM, ceW1,          2 * H_DIM, bA, H_DIM, nullptr, H_DIM, H_DIM, 0);
    gemm(bP, H_DIM, ceW1 + H_DIM,  2 * H_DIM, bA, H_DIM, ceb1,    H_DIM, H_DIM, 1);
    ln(bA, ceg1, cebe1);
    gemm(bA, H_DIM, ceW2, H_DIM, bM, H_DIM, ceb2, H_DIM, H_DIM, 0); ln(bM, ceg2, cebe2);
    // --- fusion on concat[gene, comb] via split-K ---
    gemm(bC, H_DIM, fuW1,          2 * H_DIM, bA, H_DIM, nullptr, H_DIM, H_DIM, 0);
    gemm(bM, H_DIM, fuW1 + H_DIM,  2 * H_DIM, bA, H_DIM, fub1,    H_DIM, H_DIM, 1);
    ln(bA, fug1, fube1);
    gemm(bA, H_DIM, fuW2, H_DIM, bX, H_DIM, fub2, H_DIM, H_DIM, 0); ln(bX, fug2, fube2);
    // --- output head ---
    gemm(bX, H_DIM, oW1, H_DIM, bA, H_DIM, ob1, H_DIM, H_DIM, 0); ln(bA, og1, obe1);
    gemm(bA, H_DIM, oW2, H_DIM, out, D_IN, ob2, D_IN, H_DIM, 0);
}
